// _IPEXAttention_50972671869036
// MI455X (gfx1250) — matrix operand order repaired and checked by reading, not compiled, not run on hardware
//
#include <hip/hip_runtime.h>
#include <hip/hip_bf16.h>

typedef unsigned short u16;
typedef __attribute__((ext_vector_type(16))) __bf16 v16bf;
typedef __attribute__((ext_vector_type(8)))  float  v8f;

#define HIDDEN 4096
#define NHEADS 32
#define HDIM   128
#define SEQ    2048
#define BATCH  2
#define KDIM   4096
#define NDIM   (3*4096)
#define MDIM   (BATCH*SEQ)

union Frag {
    v16bf  v;
    float4 f4[2];
    u16    u[16];
};

__device__ __forceinline__ u16 f2bf(float f) {
    unsigned u = __float_as_uint(f);
    unsigned r = 0x7FFFu + ((u >> 16) & 1u);
    return (u16)((u + r) >> 16);
}
__device__ __forceinline__ float bf2f(u16 u) {
    return __uint_as_float(((unsigned)u) << 16);
}

// LDS byte offset of a generic pointer to __shared__ (low 32 bits).
__device__ __forceinline__ unsigned ldsoff(const void* p) {
    return (unsigned)(unsigned long long)p;
}
// gfx1250 async global->LDS copy, 16 bytes per lane. Tracked by ASYNCcnt.
__device__ __forceinline__ void async_b128(unsigned lds, const void* g) {
    asm volatile("global_load_async_to_lds_b128 %0, %1, off"
                 :: "v"(lds), "v"((unsigned long long)g) : "memory");
}
__device__ __forceinline__ void wait_async0() {
    asm volatile("s_wait_asynccnt 0x0" ::: "memory");
}
__device__ __forceinline__ void wait_ds0() {
    asm volatile("s_wait_dscnt 0x0" ::: "memory");
}

// ---------------------------------------------------------------------------
// Kernel 1: fp32 -> bf16 convert
// ---------------------------------------------------------------------------
__global__ __launch_bounds__(256) void k_cvt_bf16(const float* __restrict__ in,
                                                  u16* __restrict__ out, int n4) {
    int i = blockIdx.x * 256 + threadIdx.x;
    if (i < n4) {
        float4 f = ((const float4*)in)[i];
        uint2 pk;
        pk.x = (unsigned)f2bf(f.x) | ((unsigned)f2bf(f.y) << 16);
        pk.y = (unsigned)f2bf(f.z) | ((unsigned)f2bf(f.w) << 16);
        ((uint2*)out)[i] = pk;
    }
}

// ---------------------------------------------------------------------------
// Kernel 2: transpose + convert qkv_w: W[K=4096][N=12288] f32 -> Wt[N][K] bf16
// ---------------------------------------------------------------------------
__global__ __launch_bounds__(256) void k_transpose_w(const float* __restrict__ W,
                                                     u16* __restrict__ Wt) {
    __shared__ float tile[32][33];
    int x = blockIdx.x * 32 + threadIdx.x;
    int y = blockIdx.y * 32 + threadIdx.y;
    #pragma unroll
    for (int j = 0; j < 32; j += 8)
        tile[threadIdx.y + j][threadIdx.x] = W[(size_t)(y + j) * NDIM + x];
    __syncthreads();
    int xo = blockIdx.y * 32 + threadIdx.x;
    int yo = blockIdx.x * 32 + threadIdx.y;
    #pragma unroll
    for (int j = 0; j < 32; j += 8)
        Wt[(size_t)(yo + j) * KDIM + xo] = f2bf(tile[threadIdx.x][threadIdx.y + j]);
}

// ---------------------------------------------------------------------------
// Kernel 3: QKV GEMM, block tile 128(M)x128(N), K-step 32.
// A/B slabs double-buffered in LDS via global_load_async_to_lds_b128;
// one s_barrier per K-step; fragments fed to WMMA via ds_load_b128.
// Waves: 4x2 grid of 32(M)x64(N) tiles.
// Epilogue scatters Q,K -> [b][h][s][d] bf16 ; V -> [b][h][d][s] bf16.
// ---------------------------------------------------------------------------
__global__ __launch_bounds__(256) void k_gemm_qkv(const u16* __restrict__ X,
                                                  const u16* __restrict__ Wt,
                                                  u16* __restrict__ Qb,
                                                  u16* __restrict__ Kb,
                                                  u16* __restrict__ Vt) {
    __shared__ u16 Asm[2][128 * 40];   // padded: conflict-free frag reads
    __shared__ u16 Bsm[2][128 * 40];

    const int tid  = threadIdx.x;
    const int lane = tid & 31, wave = tid >> 5;
    const int hi = lane >> 4, lo = lane & 15;

    const int blkM = (blockIdx.x / 96) * 128;
    const int blkN = (blockIdx.x % 96) * 128;
    const int waveM = (wave >> 1) * 32;
    const int waveN = (wave & 1) * 64;

    // Tile loader mapping: each thread owns one 32B chunk (16 u16) per slab.
    const int trow = tid >> 1;
    const int tck  = (tid & 1) * 16;
    const u16* gA = X  + (size_t)(blkM + trow) * KDIM + tck;
    const u16* gB = Wt + (size_t)(blkN + trow) * KDIM + tck;

    v8f acc[2][4];
    #pragma unroll
    for (int mt = 0; mt < 2; ++mt)
        #pragma unroll
        for (int nt = 0; nt < 4; ++nt)
            acc[mt][nt] = (v8f){0.f,0.f,0.f,0.f,0.f,0.f,0.f,0.f};

    // Prologue: async-stage K-slab 0 into buffer 0.
    {
        unsigned la = ldsoff(&Asm[0][trow * 40 + tck]);
        unsigned lb = ldsoff(&Bsm[0][trow * 40 + tck]);
        async_b128(la,      gA);     async_b128(la + 16, gA + 8);
        async_b128(lb,      gB);     async_b128(lb + 16, gB + 8);
    }
    wait_async0();
    __syncthreads();

    int buf = 0;
    for (int k = 0; k < KDIM; k += 32, buf ^= 1) {
        // Issue async stage of next K-slab into the other buffer.
        if (k + 32 < KDIM) {
            unsigned la = ldsoff(&Asm[buf ^ 1][trow * 40 + tck]);
            unsigned lb = ldsoff(&Bsm[buf ^ 1][trow * 40 + tck]);
            async_b128(la,      gA + k + 32);  async_b128(la + 16, gA + k + 40);
            async_b128(lb,      gB + k + 32);  async_b128(lb + 16, gB + k + 40);
            __builtin_prefetch(gA + k + 64, 0, 1);   // warm L2 one slab ahead
        }
        // Fragments from LDS (ds_load_b128) and 8 WMMAs.
        Frag a[2], b[4];
        #pragma unroll
        for (int mt = 0; mt < 2; ++mt) {
            const u16* p = &Asm[buf][(waveM + mt * 16 + lo) * 40 + hi * 8];
            a[mt].f4[0] = *(const float4*)p;
            a[mt].f4[1] = *(const float4*)(p + 16);
        }
        #pragma unroll
        for (int nt = 0; nt < 4; ++nt) {
            const u16* p = &Bsm[buf][(waveN + nt * 16 + lo) * 40 + hi * 8];
            b[nt].f4[0] = *(const float4*)p;
            b[nt].f4[1] = *(const float4*)(p + 16);
        }
        #pragma unroll
        for (int mt = 0; mt < 2; ++mt)
            #pragma unroll
            for (int nt = 0; nt < 4; ++nt)
                acc[mt][nt] = __builtin_amdgcn_wmma_f32_16x16x32_bf16(
                    false, a[mt].v, false, b[nt].v, (short)0, acc[mt][nt],
                    false, false);
        wait_async0();
        __syncthreads();
    }

    // Epilogue scatter. C layout: m = r + 8*hi, n = lo.
    #pragma unroll
    for (int mt = 0; mt < 2; ++mt) {
        #pragma unroll
        for (int nt = 0; nt < 4; ++nt) {
            int ng  = blkN + waveN + nt * 16 + lo;
            int sec = ng >> 12;          // 0=Q 1=K 2=V
            int rem = ng & 4095;
            int h   = rem >> 7;
            int d   = rem & 127;
            #pragma unroll
            for (int r = 0; r < 8; ++r) {
                int mg = blkM + waveM + mt * 16 + r + 8 * hi;
                int bb = mg >> 11;
                int s  = mg & 2047;
                u16 val = f2bf(acc[mt][nt][r]);
                size_t bh = (size_t)(bb * NHEADS + h);
                if (sec == 0)       Qb[(bh * SEQ + s) * HDIM + d] = val;
                else if (sec == 1)  Kb[(bh * SEQ + s) * HDIM + d] = val;
                else                Vt[(bh * HDIM + d) * SEQ + s] = val;
            }
        }
    }
}

// ---------------------------------------------------------------------------
// Kernel 4: in-place RoPE (rotate-half) on Q and K, bf16.
// ---------------------------------------------------------------------------
__global__ __launch_bounds__(256) void k_rope(u16* __restrict__ Qb,
                                              u16* __restrict__ Kb,
                                              const int* __restrict__ pos_ids) {
    int idx = blockIdx.x * 256 + threadIdx.x;
    int d = idx & 63;       int t = idx >> 6;
    int s = t & 2047;       t >>= 11;
    int bh = t & 63;        int which = t >> 6;
    u16* base = (which ? Kb : Qb) + ((size_t)bh * SEQ + s) * HDIM;
    int pos = pos_ids[(bh >> 5) * SEQ + s];
    float inv = __expf((float)d * (-2.0f * 9.210340371976184f / 128.0f));
    float ang = (float)pos * inv;
    float sn, cs;
    __sincosf(ang, &sn, &cs);
    float x1 = bf2f(base[d]);
    float x2 = bf2f(base[d + 64]);
    base[d]      = f2bf(x1 * cs - x2 * sn);
    base[d + 64] = f2bf(x2 * cs + x1 * sn);
}

// ---------------------------------------------------------------------------
// Kernel 5: causal flash attention with LDS-staged K/V tiles.
// Grid: 64 (b,h) x 16 q-chunks of 128 rows; 8 waves; 16 q-rows/wave.
// Per key-tile of 32: K (32x128) and V (128x32, pre-transposed) are staged
// into double-buffered LDS via async-to-LDS, shared by all 8 waves.
// Block-uniform loop; per-wave causal predication (wave-uniform => EXEC full
// around WMMA). P staged via padded LDS with s_wait_dscnt ordering.
// ---------------------------------------------------------------------------
__global__ __launch_bounds__(256) void k_attn(const u16* __restrict__ Qb,
                                              const u16* __restrict__ Kb,
                                              const u16* __restrict__ Vt,
                                              const float* __restrict__ mask,
                                              float* __restrict__ out) {
    __shared__ u16 Ksm[2][32 * 136];   // key-major, pad: 4r mod 64 distinct
    __shared__ u16 Vsm[2][128 * 40];   // d-major,  pad: 20r mod 64 distinct
    __shared__ u16 Psm[8][16 * 40];    // per-wave P tile

    const int tid  = threadIdx.x;
    const int lane = tid & 31, wave = tid >> 5;
    const int hi = lane >> 4, lo = lane & 15;

    int bh = blockIdx.x >> 4;
    int qc = blockIdx.x & 15;
    int b  = bh >> 5, h = bh & 31;
    int qBase = qc * 128 + wave * 16;

    const u16*  Qr = Qb + (size_t)bh * SEQ * HDIM;
    const u16*  Kr = Kb + (size_t)bh * SEQ * HDIM;
    const u16*  Vr = Vt + (size_t)bh * HDIM * SEQ;
    const float* Mr = mask + (size_t)b * SEQ * SEQ;
    const float scale = 0.08838834764831845f;
    const float NEG_BIG = -3.0e38f;

    // Resident Q fragments.
    Frag aq[4];
    #pragma unroll
    for (int dsp = 0; dsp < 4; ++dsp) {
        const u16* p = Qr + (size_t)(qBase + lo) * HDIM + dsp * 32 + hi * 8;
        aq[dsp].f4[0] = *(const float4*)p;
        aq[dsp].f4[1] = *(const float4*)(p + 16);
    }

    v8f o[8];
    #pragma unroll
    for (int nt = 0; nt < 8; ++nt)
        o[nt] = (v8f){0.f,0.f,0.f,0.f,0.f,0.f,0.f,0.f};
    float mrun[8], lrun[8], alphav[8];
    #pragma unroll
    for (int r = 0; r < 8; ++r) { mrun[r] = NEG_BIG; lrun[r] = 0.f; }

    // Tile loader mapping (per thread: 32B of K tile, 32B of V tile).
    const int krow = tid >> 3, kseg = (tid & 7) * 16;   // K: 32 x (8 segs)
    const int vrow = tid >> 1, vseg = (tid & 1) * 16;   // V: 128 x (2 segs)

    const int kcEndBlk = qc * 4 + 4;
    const int kcEndWav = ((qBase + 15) >> 5) + 1;

    // Prologue: stage key-tile 0.
    {
        const u16* gk = Kr + (size_t)krow * HDIM + kseg;
        const u16* gv = Vr + (size_t)vrow * SEQ + vseg;
        unsigned lk = ldsoff(&Ksm[0][krow * 136 + kseg]);
        unsigned lv = ldsoff(&Vsm[0][vrow * 40 + vseg]);
        async_b128(lk, gk);  async_b128(lk + 16, gk + 8);
        async_b128(lv, gv);  async_b128(lv + 16, gv + 8);
    }
    wait_async0();
    __syncthreads();

    int buf = 0;
    for (int kc = 0; kc < kcEndBlk; ++kc, buf ^= 1) {
        int k0 = kc * 32;
        if (kc + 1 < kcEndBlk) {   // stage next key-tile into other buffer
            const u16* gk = Kr + (size_t)(k0 + 32 + krow) * HDIM + kseg;
            const u16* gv = Vr + (size_t)vrow * SEQ + (k0 + 32) + vseg;
            unsigned lk = ldsoff(&Ksm[buf ^ 1][krow * 136 + kseg]);
            unsigned lv = ldsoff(&Vsm[buf ^ 1][vrow * 40 + vseg]);
            async_b128(lk, gk);  async_b128(lk + 16, gk + 8);
            async_b128(lv, gv);  async_b128(lv + 16, gv + 8);
        }

        if (kc < kcEndWav) {   // wave-uniform causal predicate
            v8f s0 = (v8f){0.f,0.f,0.f,0.f,0.f,0.f,0.f,0.f};
            v8f s1 = s0;
            #pragma unroll
            for (int dsp = 0; dsp < 4; ++dsp) {
                Frag bk0, bk1;
                const u16* p0 = &Ksm[buf][lo        * 136 + dsp * 32 + hi * 8];
                const u16* p1 = &Ksm[buf][(16 + lo) * 136 + dsp * 32 + hi * 8];
                bk0.f4[0] = *(const float4*)p0;  bk0.f4[1] = *(const float4*)(p0 + 16);
                bk1.f4[0] = *(const float4*)p1;  bk1.f4[1] = *(const float4*)(p1 + 16);
                s0 = __builtin_amdgcn_wmma_f32_16x16x32_bf16(
                    false, aq[dsp].v, false, bk0.v, (short)0, s0, false, false);
                s1 = __builtin_amdgcn_wmma_f32_16x16x32_bf16(
                    false, aq[dsp].v, false, bk1.v, (short)0, s1, false, false);
            }

            #pragma unroll
            for (int r = 0; r < 8; ++r) {
                int mg = qBase + r + 8 * hi;
                int n0 = k0 + lo, n1 = n0 + 16;
                float v0 = s0[r] * scale + Mr[(size_t)mg * SEQ + n0];
                float v1 = s1[r] * scale + Mr[(size_t)mg * SEQ + n1];
                if (n0 > mg) v0 = NEG_BIG;
                if (n1 > mg) v1 = NEG_BIG;

                float tm = fmaxf(v0, v1);
                tm = fmaxf(tm, __shfl_xor(tm, 1, 32));
                tm = fmaxf(tm, __shfl_xor(tm, 2, 32));
                tm = fmaxf(tm, __shfl_xor(tm, 4, 32));
                tm = fmaxf(tm, __shfl_xor(tm, 8, 32));
                float nm = fmaxf(mrun[r], tm);
                float al = __expf(mrun[r] - nm);
                mrun[r] = nm;
                alphav[r] = al;

                float e0 = __expf(v0 - nm);
                float e1 = __expf(v1 - nm);
                float ts = e0 + e1;
                ts += __shfl_xor(ts, 1, 32);
                ts += __shfl_xor(ts, 2, 32);
                ts += __shfl_xor(ts, 4, 32);
                ts += __shfl_xor(ts, 8, 32);
                lrun[r] = lrun[r] * al + ts;

                int row = r + 8 * hi;
                Psm[wave][row * 40 + lo]      = f2bf(e0);
                Psm[wave][row * 40 + 16 + lo] = f2bf(e1);
            }

            #pragma unroll
            for (int nt = 0; nt < 8; ++nt)
                #pragma unroll
                for (int r = 0; r < 8; ++r)
                    o[nt][r] *= alphav[r];

            wait_ds0();
            Frag ap;
            const u16* pp = &Psm[wave][lo * 40 + hi * 8];
            ap.f4[0] = *(const float4*)pp;
            ap.f4[1] = *(const float4*)(pp + 16);

            #pragma unroll
            for (int nt = 0; nt < 8; ++nt) {
                Frag bv;
                const u16* pv = &Vsm[buf][(nt * 16 + lo) * 40 + hi * 8];
                bv.f4[0] = *(const float4*)pv;
                bv.f4[1] = *(const float4*)(pv + 16);
                o[nt] = __builtin_amdgcn_wmma_f32_16x16x32_bf16(
                    false, ap.v, false, bv.v, (short)0, o[nt], false, false);
            }
            wait_ds0();
        }

        wait_async0();
        __syncthreads();
    }

    float inv[8];
    #pragma unroll
    for (int r = 0; r < 8; ++r) inv[r] = 1.0f / lrun[r];
    #pragma unroll
    for (int nt = 0; nt < 8; ++nt) {
        int d = nt * 16 + lo;
        #pragma unroll
        for (int r = 0; r < 8; ++r) {
            int mg = qBase + r + 8 * hi;
            out[((size_t)b * SEQ + mg) * HIDDEN + h * HDIM + d] = o[nt][r] * inv[r];
        }
    }
}

// ---------------------------------------------------------------------------
extern "C" void kernel_launch(void* const* d_in, const int* in_sizes, int n_in,
                              void* d_out, int out_size, void* d_ws, size_t ws_size,
                              hipStream_t stream) {
    const float* hs    = (const float*)d_in[0];
    const float* amask = (const float*)d_in[1];
    const float* qkv_w = (const float*)d_in[2];
    const int*   pos   = (const int*)d_in[3];
    float* out = (float*)d_out;

    char* ws = (char*)d_ws;
    u16* Xb = (u16*)(ws);
    u16* Wt = (u16*)(ws + (size_t)32  * 1024 * 1024);
    u16* Qb = (u16*)(ws + (size_t)128 * 1024 * 1024);
    u16* Kb = (u16*)(ws + (size_t)160 * 1024 * 1024);
    u16* Vt = (u16*)(ws + (size_t)192 * 1024 * 1024);

    k_cvt_bf16<<<(MDIM * KDIM / 4) / 256, 256, 0, stream>>>(hs, Xb, MDIM * KDIM / 4);
    k_transpose_w<<<dim3(NDIM / 32, KDIM / 32), dim3(32, 8), 0, stream>>>(qkv_w, Wt);
    k_gemm_qkv<<<32 * 96, 256, 0, stream>>>(Xb, Wt, Qb, Kb, Vt);
    k_rope<<<(2 * BATCH * NHEADS * SEQ * 64) / 256, 256, 0, stream>>>(Qb, Kb, pos);
    k_attn<<<64 * 16, 256, 0, stream>>>(Qb, Kb, Vt, amask, out);
}
